// GCN_13838384627834
// MI455X (gfx1250) — matrix operand order repaired and checked by reading, compiled, not run on hardware
//
#include <hip/hip_runtime.h>
#include <hip/hip_bf16.h>

#define N_NODES  100000
#define N_EDGES  400000
#define N_GRAPHS 64

typedef __bf16        v16bf __attribute__((ext_vector_type(16)));
typedef float         v8f   __attribute__((ext_vector_type(8)));
typedef float         v4f   __attribute__((ext_vector_type(4)));
typedef unsigned int  v4u   __attribute__((ext_vector_type(4)));

__device__ __forceinline__ unsigned short f2bf(float x) {
    unsigned int u = __float_as_uint(x);
    u += 0x7FFFu + ((u >> 16) & 1u);     // round-to-nearest-even
    return (unsigned short)(u >> 16);
}

// ---------------------------------------------------------------- degree
__global__ void fill1_k(float* p, int n) {
    int i = blockIdx.x * blockDim.x + threadIdx.x;
    if (i < n) p[i] = 1.0f;
}
__global__ void deg_acc_k(const int* __restrict__ dst, float* __restrict__ deg, int E) {
    int i = blockIdx.x * blockDim.x + threadIdx.x;
    if (i < E) atomicAdd(&deg[dst[i]], 1.0f);
}
__global__ void rsq_k(float* p, int n) {
    int i = blockIdx.x * blockDim.x + threadIdx.x;
    if (i < n) p[i] = rsqrtf(p[i]);
}

// ---------------------------------------------------------------- converts
// W[k][n] f32 -> Wt[n][k] bf16 (transposed so B tiles are K-contiguous)
__global__ void wconv_k(const float* __restrict__ W, unsigned short* __restrict__ Wt,
                        int K, int Nc) {
    int t = blockIdx.x * blockDim.x + threadIdx.x;
    if (t >= K * Nc) return;
    int k = t / Nc, n = t % Nc;
    Wt[(size_t)n * K + k] = f2bf(W[t]);
}
// activations f32 -> bf16, same layout (done once per layer)
__global__ void aconv_k(const float* __restrict__ in, unsigned short* __restrict__ ob, int total) {
    int t = blockIdx.x * blockDim.x + threadIdx.x;
    if (t < total) ob[t] = f2bf(in[t]);
}

// ---------------------------------------------------------------- WMMA GEMM
// C[M,Nc] = A[M,K](bf16) @ Wt[Nc,K](bf16)^T ; block tile 128x64, 8 waves of 32x32,
// double-buffered LDS stages filled by GLOBAL_LOAD_ASYNC_TO_LDS_B128 (ASYNCcnt).
#define AS_STRIDE 40                 // 32 + 8 pad bf16; 80 B rows, 16 B aligned
#define A_STAGE   (128 * AS_STRIDE)  // ushorts
#define B_STAGE   (64  * AS_STRIDE)

__global__ __launch_bounds__(256)
void gemm_wmma_k(const unsigned short* __restrict__ Abf,
                 const unsigned short* __restrict__ Wt,
                 float* __restrict__ C, int M, int K, int Nc) {
    __shared__ unsigned short As[2 * A_STAGE];
    __shared__ unsigned short Bs[2 * B_STAGE];

    const int tid  = threadIdx.x;
    const int lane = tid & 31;
    const int wave = tid >> 5;
    const int half = (lane >> 4) & 1;
    const int l16  = lane & 15;
    const int wm   = wave & 3;
    const int wn   = wave >> 2;
    const int m_base = blockIdx.x * 128;
    const int n_base = blockIdx.y * 64;

    const unsigned long long abase = (unsigned long long)Abf;
    const unsigned long long bbase = (unsigned long long)Wt;

    // per-thread staging coordinates
    const int a_q   = tid & 3;           // 16 B quarter of a 64 B row
    const int a_row0 = tid >> 2;         // rows 0..63  (j=0)
    const int a_row1 = a_row0 + 64;      // rows 64..127 (j=1)
    const int b_n   = tid >> 2;          // 0..63
    const int b_p   = tid & 3;

    int arg0 = m_base + a_row0; if (arg0 > M - 1) arg0 = M - 1;   // clamp tail
    int arg1 = m_base + a_row1; if (arg1 > M - 1) arg1 = M - 1;

    // issue 3 async b128 loads (2 for A, 1 for B) into LDS stage
    auto issue = [&](int stage, int kk) {
        unsigned lds, voff;
        lds  = (unsigned)(uintptr_t)&As[stage * A_STAGE + a_row0 * AS_STRIDE + a_q * 8];
        voff = (unsigned)(((size_t)arg0 * K + kk + a_q * 8) * 2);
        asm volatile("global_load_async_to_lds_b128 %0, %1, %2"
                     :: "v"(lds), "v"(voff), "s"(abase) : "memory");
        lds  = (unsigned)(uintptr_t)&As[stage * A_STAGE + a_row1 * AS_STRIDE + a_q * 8];
        voff = (unsigned)(((size_t)arg1 * K + kk + a_q * 8) * 2);
        asm volatile("global_load_async_to_lds_b128 %0, %1, %2"
                     :: "v"(lds), "v"(voff), "s"(abase) : "memory");
        lds  = (unsigned)(uintptr_t)&Bs[stage * B_STAGE + b_n * AS_STRIDE + b_p * 8];
        voff = (unsigned)(((size_t)(n_base + b_n) * K + kk + b_p * 8) * 2);
        asm volatile("global_load_async_to_lds_b128 %0, %1, %2"
                     :: "v"(lds), "v"(voff), "s"(bbase) : "memory");
    };

    const v8f vzero = {0.f, 0.f, 0.f, 0.f, 0.f, 0.f, 0.f, 0.f};
    v8f acc[2][2];
    acc[0][0] = vzero; acc[0][1] = vzero; acc[1][0] = vzero; acc[1][1] = vzero;

    union Frag { v4u q[2]; v16bf v; };

    issue(0, 0);
    int stage = 0;
    for (int kk = 0; kk < K; kk += 32) {
        if (kk + 32 < K) {
            issue(stage ^ 1, kk + 32);                       // prefetch next stage
            asm volatile("s_wait_asynccnt 0x3" ::: "memory"); // own cur-stage loads done
        } else {
            asm volatile("s_wait_asynccnt 0x0" ::: "memory");
        }
        __syncthreads();   // all waves' cur-stage loads visible

        const unsigned short* Ab = &As[stage * A_STAGE];
        const unsigned short* Bb = &Bs[stage * B_STAGE];

        Frag af[2], bf[2];
        #pragma unroll
        for (int mt = 0; mt < 2; ++mt) {
            const unsigned short* ap = &Ab[(wm * 32 + mt * 16 + l16) * AS_STRIDE];
            af[mt].q[0] = *(const v4u*)(ap + half * 8);        // K 0-7   / 8-15
            af[mt].q[1] = *(const v4u*)(ap + 16 + half * 8);   // K 16-23 / 24-31
        }
        #pragma unroll
        for (int nt = 0; nt < 2; ++nt) {
            const unsigned short* bp = &Bb[(wn * 32 + nt * 16 + l16) * AS_STRIDE + half * 8];
            bf[nt].q[0] = *(const v4u*)(bp);                   // K 0-7   / 8-15
            bf[nt].q[1] = *(const v4u*)(bp + 16);              // K 16-23 / 24-31
        }
        #pragma unroll
        for (int mt = 0; mt < 2; ++mt)
            #pragma unroll
            for (int nt = 0; nt < 2; ++nt)
                acc[mt][nt] = __builtin_amdgcn_wmma_f32_16x16x32_bf16(
                    false, af[mt].v, false, bf[nt].v,
                    (short)0, acc[mt][nt], false, false);

        __syncthreads();   // all waves done reading; stage can be overwritten
        stage ^= 1;
    }

    #pragma unroll
    for (int mt = 0; mt < 2; ++mt)
        #pragma unroll
        for (int nt = 0; nt < 2; ++nt)
            #pragma unroll
            for (int g = 0; g < 8; ++g) {
                int row = m_base + wm * 32 + mt * 16 + half * 8 + g;
                int col = n_base + wn * 32 + nt * 16 + l16;
                if (row < M) C[(size_t)row * Nc + col] = acc[mt][nt][g];
            }
}

// ---------------------------------------------------------------- agg = h*dinv^2 + b
__global__ void init_agg_k(const float* __restrict__ h, const float* __restrict__ dinv,
                           const float* __restrict__ b, float* __restrict__ agg,
                           int Nc, int total) {
    int t = blockIdx.x * blockDim.x + threadIdx.x;
    if (t >= total) return;
    int i = t / Nc, f = t % Nc;
    float d = dinv[i];
    agg[t] = h[t] * d * d + b[f];
}

// ---------------------------------------------------------------- edge scatter
__global__ void scatter_k(const int* __restrict__ src, const int* __restrict__ dst,
                          const float* __restrict__ dinv, const float* __restrict__ h,
                          float* __restrict__ agg, int Nc) {
    int t = blockIdx.x * blockDim.x + threadIdx.x;
    int per = Nc >> 2;
    int e = t / per;
    if (e >= N_EDGES) return;
    int f = (t - e * per) << 2;
    int s = src[e], d = dst[e];
    float nrm = dinv[s] * dinv[d];
    v4f hv = *(const v4f*)(h + (size_t)s * Nc + f);
    float* ap = agg + (size_t)d * Nc + f;
    atomicAdd(ap + 0, hv[0] * nrm);
    atomicAdd(ap + 1, hv[1] * nrm);
    atomicAdd(ap + 2, hv[2] * nrm);
    atomicAdd(ap + 3, hv[3] * nrm);
}

__global__ void relu_k(float* p, int n) {
    int i = blockIdx.x * blockDim.x + threadIdx.x;
    if (i < n) p[i] = fmaxf(p[i], 0.f);
}

// ---------------------------------------------------------------- pooling + head
__global__ void zero_k(float* p, int n) {
    int i = blockIdx.x * blockDim.x + threadIdx.x;
    if (i < n) p[i] = 0.f;
}
__global__ void pool_k(const float* __restrict__ h, const int* __restrict__ batch,
                       float* __restrict__ sums, float* __restrict__ cnts) {
    int t = blockIdx.x * blockDim.x + threadIdx.x;
    if (t >= N_NODES * 64) return;
    int i = t >> 6, f = t & 63;
    int g = batch[i];
    atomicAdd(&sums[g * 64 + f], h[t]);
    if (f == 0) atomicAdd(&cnts[g], 1.0f);
}
__global__ void head_k(const float* __restrict__ sums, const float* __restrict__ cnts,
                       const float* __restrict__ Wl1, const float* __restrict__ bl1,
                       const float* __restrict__ Wl, const float* __restrict__ bl,
                       float* __restrict__ out) {
    int g = threadIdx.x;
    if (g >= N_GRAPHS) return;
    float c = fmaxf(cnts[g], 1.0f);
    float hid[32];
    #pragma unroll
    for (int j = 0; j < 32; ++j) {
        float s = bl1[j];
        for (int k = 0; k < 64; ++k)
            s += (sums[g * 64 + k] / c) * Wl1[k * 32 + j];
        hid[j] = fmaxf(s, 0.f);
    }
    #pragma unroll
    for (int o = 0; o < 2; ++o) {
        float s = bl[o];
        for (int j = 0; j < 32; ++j) s += hid[j] * Wl[j * 2 + o];
        out[g * 2 + o] = s;
    }
}

// ---------------------------------------------------------------- launcher
static inline int cdiv(int a, int b) { return (a + b - 1) / b; }

extern "C" void kernel_launch(void* const* d_in, const int* in_sizes, int n_in,
                              void* d_out, int out_size, void* d_ws, size_t ws_size,
                              hipStream_t stream) {
    (void)in_sizes; (void)n_in; (void)out_size; (void)ws_size;

    const float* x     = (const float*)d_in[0];
    const int*   ei    = (const int*)d_in[1];
    const int*   src   = ei;
    const int*   dst   = ei + N_EDGES;
    const int*   batch = (const int*)d_in[2];
    const float* W[4]  = {(const float*)d_in[3], (const float*)d_in[5],
                          (const float*)d_in[7], (const float*)d_in[9]};
    const float* b[4]  = {(const float*)d_in[4], (const float*)d_in[6],
                          (const float*)d_in[8], (const float*)d_in[10]};
    const float* Wl1   = (const float*)d_in[11];
    const float* bl1   = (const float*)d_in[12];
    const float* Wl    = (const float*)d_in[13];
    const float* bl    = (const float*)d_in[14];
    float* out = (float*)d_out;

    // workspace layout
    float* buf0 = (float*)d_ws;                         // h   [N,512] max
    float* buf1 = buf0 + (size_t)N_NODES * 512;         // agg/out [N,512] max
    float* dinv = buf1 + (size_t)N_NODES * 512;         // [N]
    float* sums = dinv + N_NODES;                       // [64,64]
    float* cnts = sums + N_GRAPHS * 64;                 // [64]
    unsigned short* wt  = (unsigned short*)(cnts + N_GRAPHS); // [512*512] bf16
    unsigned short* abf = wt + 512 * 512;                     // [N,512] bf16

    fill1_k<<<cdiv(N_NODES, 256), 256, 0, stream>>>(dinv, N_NODES);
    deg_acc_k<<<cdiv(N_EDGES, 256), 256, 0, stream>>>(dst, dinv, N_EDGES);
    rsq_k<<<cdiv(N_NODES, 256), 256, 0, stream>>>(dinv, N_NODES);

    const int dk[4] = {512, 512, 256, 128};
    const int dn[4] = {512, 256, 128, 64};
    const float* in = x;
    for (int l = 0; l < 4; ++l) {
        int K = dk[l], Nc = dn[l];
        aconv_k<<<cdiv(N_NODES * K, 256), 256, 0, stream>>>(in, abf, N_NODES * K);
        wconv_k<<<cdiv(K * Nc, 256), 256, 0, stream>>>(W[l], wt, K, Nc);
        dim3 grid(cdiv(N_NODES, 128), Nc / 64);
        gemm_wmma_k<<<grid, 256, 0, stream>>>(abf, wt, buf0, N_NODES, K, Nc);
        int tot = N_NODES * Nc;
        init_agg_k<<<cdiv(tot, 256), 256, 0, stream>>>(buf0, dinv, b[l], buf1, Nc, tot);
        scatter_k<<<cdiv(N_EDGES * (Nc / 4), 256), 256, 0, stream>>>(src, dst, dinv, buf0, buf1, Nc);
        if (l < 3) relu_k<<<cdiv(tot, 256), 256, 0, stream>>>(buf1, tot);
        in = buf1;
    }

    zero_k<<<cdiv(N_GRAPHS * 64 + N_GRAPHS, 256), 256, 0, stream>>>(sums, N_GRAPHS * 64 + N_GRAPHS);
    pool_k<<<cdiv(N_NODES * 64, 256), 256, 0, stream>>>(buf1, batch, sums, cnts);
    head_k<<<1, 64, 0, stream>>>(sums, cnts, Wl1, bl1, Wl, bl, out);
}
